// OuterProductMean_2370821947566
// MI455X (gfx1250) — matrix operand order repaired and checked by reading, not compiled, not run on hardware
//
#include <hip/hip_runtime.h>

// ---------------------------------------------------------------------------
// OuterProductMean for MI455X (gfx1250), wave32, WMMA bf16 path.
//   kernel 0: Wo f32 -> bf16 WMMA B-fragments
//   kernel 1: LayerNorm + left/right projections -> bf16 WMMA A/B fragments
//   kernel 2: fused C = A*B (per 16x16 (l,m) tile) -> out = vec(C)*Wo + bo
//             panel staging via GLOBAL_LOAD_ASYNC_TO_LDS_B128 (ASYNCcnt)
// ---------------------------------------------------------------------------

typedef __bf16 v16bf __attribute__((ext_vector_type(16)));
typedef __bf16 v8bf  __attribute__((ext_vector_type(8)));
typedef float  v8f   __attribute__((ext_vector_type(8)));

#define L_SEQ  512
#define N_DEP  256
#define DMSA   256
#define DHID   16
#define DPAIR  128
#define KSTEP  8          // 256 / 32 (bf16 WMMA k=32)
#define FRAG_E 512        // elements per fragment: 32 lanes * 16 bf16

// --------------------------------------------------------------------------
// Kernel 0: convert Wo [256,128] f32 row-major -> bf16 B-fragment layout
// WoF[nt(8)][ks(8)][lane(32)][elem(16)]; B layout: lanes 0-15 = col j, K 0..7 / 16..23
// in elems 0..7 / 8..15; lanes 16-31 = col j, K 8..15 / 24..31.
// --------------------------------------------------------------------------
__global__ __launch_bounds__(256) void k_cvt_wo(const float* __restrict__ Wo,
                                                __bf16* __restrict__ WoF) {
  int idx  = blockIdx.x * 256 + threadIdx.x;   // 0..32767
  int elem = idx & 15;
  int lane = (idx >> 4) & 31;
  int ks   = (idx >> 9) & 7;
  int nt   = idx >> 12;
  int j  = lane & 15;
  int kp = ((lane >> 4) << 3) + (elem & 7) + ((elem >> 3) << 4);          // k within 32-step
  int k  = ks * 32 + kp;
  int n  = nt * 16 + j;
  WoF[idx] = (__bf16)Wo[k * DPAIR + n];
}

// --------------------------------------------------------------------------
// Kernel 1: one wave32 per (s,l) row. LayerNorm(256) then two 256->16
// projections; results scattered directly into WMMA fragment layout.
// A fragment (16x32, M=i hidden, K=s'): lane = i + 16*((s'/8)&1),
//   elem = (s'%8) + 8*(s'/16).
// B fragment (32x16, K=s', N=j hidden): lane = j + 16*((s'/8)&1), elem = (s'%8) + 8*(s'/16).
// --------------------------------------------------------------------------
__global__ __launch_bounds__(256) void k_ln_proj(
    const float* __restrict__ msa, const float* __restrict__ gamma,
    const float* __restrict__ beta,
    const float* __restrict__ Wl, const float* __restrict__ bl,
    const float* __restrict__ Wr, const float* __restrict__ br,
    __bf16* __restrict__ Afrag, __bf16* __restrict__ Bfrag) {
  __shared__ float sWl[DMSA * DHID];
  __shared__ float sWr[DMSA * DHID];
  __shared__ float sG[DMSA], sBt[DMSA];
  for (int i = threadIdx.x; i < DMSA * DHID; i += 256) {
    sWl[i] = Wl[i];
    sWr[i] = Wr[i];
  }
  if (threadIdx.x < DMSA) {
    sG[threadIdx.x]  = gamma[threadIdx.x];
    sBt[threadIdx.x] = beta[threadIdx.x];
  }
  __syncthreads();

  const int lane = threadIdx.x & 31;
  const int wv   = threadIdx.x >> 5;
  const int row  = blockIdx.x * 8 + wv;   // row = s*L + l, 131072 rows
  const int s    = row >> 9;
  const int l    = row & (L_SEQ - 1);
  const float* xr = msa + (size_t)row * DMSA;

  float4 v0 = *(const float4*)(xr + lane * 4);
  float4 v1 = *(const float4*)(xr + 128 + lane * 4);
  float x[8] = {v0.x, v0.y, v0.z, v0.w, v1.x, v1.y, v1.z, v1.w};

  float sum = 0.f, sq = 0.f;
#pragma unroll
  for (int e = 0; e < 8; ++e) { sum += x[e]; sq += x[e] * x[e]; }
#pragma unroll
  for (int off = 16; off > 0; off >>= 1) {
    sum += __shfl_xor(sum, off, 32);
    sq  += __shfl_xor(sq, off, 32);
  }
  const float mu  = sum * (1.f / 256.f);
  const float var = sq * (1.f / 256.f) - mu * mu;
  const float rs  = rsqrtf(var + 1e-5f);

  float xn[8];
#pragma unroll
  for (int e = 0; e < 8; ++e) {
    int c = (e < 4) ? (lane * 4 + e) : (128 + lane * 4 + (e - 4));
    xn[e] = (x[e] - mu) * rs * sG[c] + sBt[c];
  }

  float accL[16], accR[16];
#pragma unroll
  for (int h = 0; h < 16; ++h) { accL[h] = 0.f; accR[h] = 0.f; }
#pragma unroll
  for (int e = 0; e < 8; ++e) {
    int c = (e < 4) ? (lane * 4 + e) : (128 + lane * 4 + (e - 4));
    const float4* wl4 = (const float4*)(sWl + c * DHID);
    const float4* wr4 = (const float4*)(sWr + c * DHID);
#pragma unroll
    for (int q = 0; q < 4; ++q) {
      float4 a = wl4[q], b = wr4[q];
      accL[q * 4 + 0] += xn[e] * a.x; accL[q * 4 + 1] += xn[e] * a.y;
      accL[q * 4 + 2] += xn[e] * a.z; accL[q * 4 + 3] += xn[e] * a.w;
      accR[q * 4 + 0] += xn[e] * b.x; accR[q * 4 + 1] += xn[e] * b.y;
      accR[q * 4 + 2] += xn[e] * b.z; accR[q * 4 + 3] += xn[e] * b.w;
    }
  }

  // Cross-lane reduce; lane (h mod 16) in each half keeps sum for output h.
  float outL = 0.f, outR = 0.f;
#pragma unroll
  for (int h = 0; h < 16; ++h) {
    float a = accL[h], b = accR[h];
#pragma unroll
    for (int off = 16; off > 0; off >>= 1) {
      a += __shfl_xor(a, off, 32);
      b += __shfl_xor(b, off, 32);
    }
    if ((lane & 15) == h) { outL = a; outR = b; }
  }
  outL += bl[lane & 15];
  outR = (outR + br[lane & 15]) * (1.f / 256.f);   // /N folded here

  const int sp = s & 31;
  const int ks = s >> 5;
  const size_t base = ((size_t)(l * KSTEP + ks) * 32 + lane) * 16;
  if ((lane >> 4) == ((sp >> 3) & 1))
    Afrag[base + ((sp & 7) + ((sp >> 4) << 3))] = (__bf16)outL;
  if ((lane >> 4) == ((sp >> 3) & 1))
    Bfrag[base + ((sp & 7) + ((sp >> 4) << 3))] = (__bf16)outR;
}

// --------------------------------------------------------------------------
// Kernel 2: block = 4 l-tiles x 4 m-tiles (16 (l,m) pairs), 256 thr = 8 waves.
// Phase 0: async-stage A/B fragment panels to LDS (64KB) with
//          global_load_async_to_lds_b128 (no VGPR round-trip; ASYNCcnt).
// Phase 1: each wave WMMAs 2 tiles over K=256 (8 x wmma_f32_16x16x32_bf16).
// Phase 2: tiles -> LDS V[16 pairs][256] bf16 (reusing sA region), then each
//          wave computes one 16-wide column group of V[16,256] @ Wo[256,128].
// --------------------------------------------------------------------------
__global__ __launch_bounds__(256) void k_opm_gemm(
    const __bf16* __restrict__ Afrag, const __bf16* __restrict__ Bfrag,
    const __bf16* __restrict__ WoF, const float* __restrict__ bo,
    float* __restrict__ out) {
  __shared__ __align__(16) char smem[64 * 1024];
  char* sA = smem;                 // 32KB: 4 ltiles * 8 ksteps * 1KB frag
  char* sB = smem + 32 * 1024;     // 32KB
  __bf16* sV = (__bf16*)smem;      // 8KB, reuses sA region after barrier

  const int tid  = threadIdx.x;
  const int lane = tid & 31;
  const int wv   = tid >> 5;
  const int l0   = blockIdx.x * 4;
  const int m0   = blockIdx.y * 4;

  {
    // Async global->LDS staging of the two 32KB fragment panels.
    // Generic pointers to __shared__ carry the DS byte offset in their low
    // 32 bits (flat->LDS mapping truncates to addr[31:0]), so the truncated
    // pointer is a valid VDST operand for the async-to-LDS instructions.
    const uint4* gA = (const uint4*)Afrag + (size_t)l0 * (KSTEP * 64);
    const uint4* gB = (const uint4*)Bfrag + (size_t)m0 * (KSTEP * 64);
    unsigned ldsA = (unsigned)(uintptr_t)sA + (unsigned)tid * 16u;
    unsigned ldsB = (unsigned)(uintptr_t)sB + (unsigned)tid * 16u;
#pragma unroll
    for (int i = 0; i < 8; ++i) {
      unsigned long long ga = (unsigned long long)(uintptr_t)(gA + tid + i * 256);
      unsigned long long gb = (unsigned long long)(uintptr_t)(gB + tid + i * 256);
      asm volatile("global_load_async_to_lds_b128 %0, %1, off"
                   :: "v"(ldsA + (unsigned)i * 4096u), "v"(ga) : "memory");
      asm volatile("global_load_async_to_lds_b128 %0, %1, off"
                   :: "v"(ldsB + (unsigned)i * 4096u), "v"(gb) : "memory");
    }
    asm volatile("s_wait_asynccnt 0x0" ::: "memory");
  }
  __syncthreads();

  const int lt  = wv >> 1;
  const int mt0 = (wv & 1) * 2;
  const int mt1 = mt0 + 1;

  v8f c0 = {};
  v8f c1 = {};
#pragma unroll
  for (int ks = 0; ks < KSTEP; ++ks) {
    v16bf a  = *(const v16bf*)(sA + ((lt  * KSTEP + ks) * 32 + lane) * 32);
    v16bf b0 = *(const v16bf*)(sB + ((mt0 * KSTEP + ks) * 32 + lane) * 32);
    v16bf b1 = *(const v16bf*)(sB + ((mt1 * KSTEP + ks) * 32 + lane) * 32);
    c0 = __builtin_amdgcn_wmma_f32_16x16x32_bf16(false, a, false, b0,
                                                 (short)0, c0, false, false);
    c1 = __builtin_amdgcn_wmma_f32_16x16x32_bf16(false, a, false, b1,
                                                 (short)0, c1, false, false);
  }
  __syncthreads();  // done reading sA/sB; reuse region for V staging

  {  // C tile layout: VGPR r holds row M = r + 8*(lane/16), col N = lane%16
    const int i_hi = (lane >> 4) << 3;
    const int j    = lane & 15;
    const int p0   = lt * 4 + mt0;
    const int p1   = lt * 4 + mt1;
#pragma unroll
    for (int r = 0; r < 8; ++r) {
      sV[p0 * 256 + (r + i_hi) * 16 + j] = (__bf16)c0[r];
      sV[p1 * 256 + (r + i_hi) * 16 + j] = (__bf16)c1[r];
    }
  }
  __syncthreads();

  // Phase 2: D[16 pairs, 128] = V[16,256] @ Wo[256,128]; wave wv -> ntile wv
  const int nt   = wv;
  const int p    = lane & 15;        // A-fragment row (pair) for this lane
  const int half = lane >> 4;
  const float bias = bo[nt * 16 + (lane & 15)];

  v8f d = {};
#pragma unroll
  for (int ks = 0; ks < KSTEP; ++ks) {
    // A-fragment: elems 0..7 = k' half*8.., elems 8..15 = k' 16+half*8..
    v8bf lo = *(const v8bf*)(sV + p * 256 + ks * 32 + half * 8);
    v8bf hi = *(const v8bf*)(sV + p * 256 + ks * 32 + 16 + half * 8);
    v16bf av;
#pragma unroll
    for (int e = 0; e < 8; ++e) { av[e] = lo[e]; av[e + 8] = hi[e]; }
    v16bf bw = *(const v16bf*)(WoF + ((size_t)(nt * KSTEP + ks) * 32 + lane) * 16);
    d = __builtin_amdgcn_wmma_f32_16x16x32_bf16(false, av, false, bw,
                                                (short)0, d, false, false);
  }

#pragma unroll
  for (int r = 0; r < 8; ++r) {
    int pr = r + (half << 3);              // pair index (D-matrix row)
    int l  = l0 + (pr >> 2);
    int m  = m0 + (pr & 3);
    out[(((size_t)l * L_SEQ) + m) * DPAIR + nt * 16 + (lane & 15)] = d[r] + bias;
  }
}

// --------------------------------------------------------------------------
extern "C" void kernel_launch(void* const* d_in, const int* in_sizes, int n_in,
                              void* d_out, int out_size, void* d_ws,
                              size_t ws_size, hipStream_t stream) {
  (void)in_sizes; (void)n_in; (void)out_size; (void)ws_size;
  const float* msa   = (const float*)d_in[0];
  const float* gamma = (const float*)d_in[1];
  const float* beta  = (const float*)d_in[2];
  const float* Wl    = (const float*)d_in[3];
  const float* bl    = (const float*)d_in[4];
  const float* Wr    = (const float*)d_in[5];
  const float* br    = (const float*)d_in[6];
  const float* Wo    = (const float*)d_in[7];
  const float* bo    = (const float*)d_in[8];

  char* ws = (char*)d_ws;
  __bf16* Afrag = (__bf16*)(ws);                     // 4 MB
  __bf16* Bfrag = (__bf16*)(ws + (4u << 20));        // 4 MB
  __bf16* WoF   = (__bf16*)(ws + (8u << 20));        // 64 KB
  float* out = (float*)d_out;

  k_cvt_wo<<<128, 256, 0, stream>>>(Wo, WoF);
  k_ln_proj<<<16384, 256, 0, stream>>>(msa, gamma, beta, Wl, bl, Wr, br,
                                       Afrag, Bfrag);
  k_opm_gemm<<<dim3(128, 128), 256, 0, stream>>>(Afrag, Bfrag, WoF, bo, out);
}
